// StochasticDurationPredictor_1537598292430
// MI455X (gfx1250) — matrix operand order repaired and checked by reading, not compiled, not run on hardware
//
#include <hip/hip_runtime.h>
#include <hip/hip_bf16.h>
#include <math.h>

// ---------------------------------------------------------------------------
// StochasticDurationPredictor forward for gfx1250 (MI455X).
// Channel-mixing 1x1 convs run as bf16 WMMA GEMMs (v_wmma_f32_16x16x32_bf16,
// f32 accumulate).  All f32->bf16 conversion happens in producer kernels:
//  - weights pre-packed per GEMM into the exact A-fragment VGPR layout
//    (contiguous 32 B per lane per k-step -> b128 loads, no guards)
//  - activations staged as k-pair-interleaved bf16 (one dword == one B-frag
//    dword), halving GEMM read bandwidth vs f32.
// Each wave holds all 6 A fragments in registers and sweeps 8 column tiles.
// GEMM store path is templated: unguarded when M is a multiple of 16.
//
// Parameter leaf indexing assumes jax pytree flattening: top-level dict in
// insertion order (x, x_mask, w, e_q, params), nested dicts sorted by key,
// lists in index order  ->  288 input tensors total.
// ---------------------------------------------------------------------------

#define BB 32
#define CH 192
#define TT 1024
#define NN (BB * TT)          // 32768
#define KBINS 10
#define TAILV 5.0f
#define C_LOG2PI 1.8378770664093454f
#define NT 8                  // column tiles per wave in the GEMM

typedef __attribute__((ext_vector_type(16))) __bf16     v16bf;
typedef __attribute__((ext_vector_type(8)))  float      v8f;
typedef __attribute__((ext_vector_type(8)))  unsigned   v8u;

// round-to-nearest-even f32 -> bf16, two values packed into one dword
static __device__ __forceinline__ unsigned bfpair(float x, float y) {
  unsigned ux = __builtin_bit_cast(unsigned, x);
  unsigned uy = __builtin_bit_cast(unsigned, y);
  unsigned lo = ((ux + 0x7FFFu + ((ux >> 16) & 1u)) >> 16) & 0xFFFFu;
  unsigned hi = ((uy + 0x7FFFu + ((uy >> 16) & 1u)) >> 16) << 16;
  return lo | hi;
}

static __device__ __forceinline__ float logsigf(float x) {
  return (x < 0.f) ? (x - log1pf(expf(x))) : (-log1pf(expf(-x)));
}

// Block-wide (256 thread) reduction + one atomicAdd per block.
static __device__ __forceinline__ void blockReduceAtomic(float v, float* dst) {
  __shared__ float red[256];
  __syncthreads();
  int t = threadIdx.x;
  red[t] = v;
  __syncthreads();
  for (int s = 128; s > 0; s >>= 1) {
    if (t < s) red[t] += red[t + s];
    __syncthreads();
  }
  if (t == 0) atomicAdd(dst, red[0]);
}

// ---------------------------------------------------------------------------
// Weight packing: W[M][192] f32 -> A fragments, layout [mTile][s][lane][8 u32]
// (s = k-step of 32).  One thread per (tile, s, lane); rows >= M are zero.
// ---------------------------------------------------------------------------
__global__ __launch_bounds__(256) void k_pack_w(const float* __restrict__ W,
                                                unsigned* __restrict__ APK, int M) {
  int tiles = (M + 15) >> 4;
  int i = blockIdx.x * 256 + threadIdx.x;
  if (i >= tiles * 6 * 32) return;
  int lane  = i & 31;
  int s     = (i >> 5) % 6;
  int mTile = i / (6 * 32);
  int half = lane >> 4, l = lane & 15;
  int arow = mTile * 16 + l;
  int r    = (arow < M) ? arow : 0;
  float sel = (arow < M) ? 1.f : 0.f;
  unsigned* dst = APK + (size_t)i * 8;
#pragma unroll
  for (int v = 0; v < 8; ++v) {
    int kb = ((v < 4) ? (2 * v) : (16 + 2 * (v - 4))) + half * 8 + s * 32;
    float a0 = W[r * CH + kb]     * sel;
    float a1 = W[r * CH + kb + 1] * sel;
    dst[v] = bfpair(a0, a1);
  }
}

// ---------------------------------------------------------------------------
// WMMA GEMM:  Y[M][N] = W[M][192] * X[192][N] + bias[M]
//   APK : packed bf16 A fragments (see k_pack_w)
//   X2  : bf16 activations, uint X2[k/2][n] = {bf16(x[k][n]), bf16(x[k+1][n])}
// 4 waves / block; each wave: one 16-row tile, NT column tiles, A fragments
// held in registers across the column sweep.  GUARD=false when 16 | M
// (no exec-masked stores in the epilogue).
// ---------------------------------------------------------------------------
template <bool GUARD>
__global__ __launch_bounds__(128) void k_gemm(const unsigned* __restrict__ APK,
                                              const unsigned* __restrict__ X2,
                                              const float* __restrict__ bias,
                                              float* __restrict__ Y, int M) {
  int lane = threadIdx.x & 31;
  int wv   = threadIdx.x >> 5;
  int half = lane >> 4, l = lane & 15;
  int tiles = (M + 15) >> 4;
  int mTile = blockIdx.y * 4 + wv;
  if (mTile >= tiles) return;           // wave-uniform: EXEC stays full for WMMA

  v16bf a[6];
#pragma unroll
  for (int s = 0; s < 6; ++s) {
    const unsigned* ap = APK + ((size_t)((mTile * 6 + s) * 32 + lane)) * 8;
    v8u av = *(const v8u*)ap;           // 32 B contiguous -> 2x global_load_b128
    a[s] = __builtin_bit_cast(v16bf, av);
  }

  int col0 = blockIdx.x * (16 * NT) + l;
  for (int tn = 0; tn < NT; ++tn) {
    int col = col0 + tn * 16;
    v8f acc = {0.f, 0.f, 0.f, 0.f, 0.f, 0.f, 0.f, 0.f};
#pragma unroll
    for (int s = 0; s < 6; ++s) {
      v8u bv;
      int rbase = s * 16 + half * 4;    // X2 row = kk/2
#pragma unroll
      for (int v = 0; v < 8; ++v) bv[v] = X2[(rbase + v + (v >= 4 ? 4 : 0)) * NN + col];
      v16bf b = __builtin_bit_cast(v16bf, bv);
      acc = __builtin_amdgcn_wmma_f32_16x16x32_bf16(
          false, a[s], false, b, (short)0, acc, false, false);
    }
#pragma unroll
    for (int v = 0; v < 8; ++v) {
      int m = mTile * 16 + half * 8 + v;
      if (!GUARD || m < M) Y[m * NN + col] = acc[v] + bias[GUARD ? min(m, M - 1) : m];
    }
  }
}

// x [B][C][T] -> bf16 X2 staging (pre GEMM input); no f32 copy needed.
__global__ __launch_bounds__(256) void k_transpose_pack(const float* __restrict__ x,
                                                        unsigned* __restrict__ X2) {
  int i = blockIdx.x * 256 + threadIdx.x;
  if (i >= (CH / 2) * NN) return;
  int r = i / NN, n = i % NN;
  int b = n / TT, t = n % TT;
  float a0 = x[(b * CH + 2 * r) * TT + t];
  float a1 = x[(b * CH + 2 * r + 1) * TT + t];
  X2[i] = bfpair(a0, a1);
}

// f32 [192][N] -> bf16 X2 staging
__global__ __launch_bounds__(256) void k_pack_x(const float* __restrict__ X,
                                                unsigned* __restrict__ X2) {
  int i = blockIdx.x * 256 + threadIdx.x;
  if (i >= (CH / 2) * NN) return;
  int r = i / NN, n = i % NN;
  X2[i] = bfpair(X[(2 * r) * NN + n], X[(2 * r + 1) * NN + n]);
}

// Depthwise dilated conv (ks=3), input masked:  Y = dwconv(X*mask) + sep_b
__global__ __launch_bounds__(256) void k_dwconv(const float* __restrict__ X,
                                                const float* __restrict__ mask,
                                                const float* __restrict__ sw,
                                                const float* __restrict__ sb,
                                                float* __restrict__ Y, int dil) {
  int i = blockIdx.x * 256 + threadIdx.x;
  if (i >= CH * NN) return;
  int c = i / NN, n = i % NN;
  int b = n / TT, t = n % TT;
  float acc = sb[c];
#pragma unroll
  for (int j = 0; j < 3; ++j) {
    int tt = t + (j - 1) * dil;
    if (tt >= 0 && tt < TT) {
      int nn = b * TT + tt;
      acc += sw[c * 3 + j] * X[c * NN + nn] * mask[nn];
    }
  }
  Y[i] = acc;
}

// Channel LayerNorm + exact GELU -> bf16 X2 staging (feeds a GEMM only).
__global__ __launch_bounds__(256) void k_ln_gelu_pack(const float* __restrict__ X,
                                                      const float* __restrict__ g,
                                                      const float* __restrict__ be,
                                                      unsigned* __restrict__ X2) {
  int n = blockIdx.x * 256 + threadIdx.x;
  if (n >= NN) return;
  float s = 0.f, s2 = 0.f;
  for (int c = 0; c < CH; ++c) {
    float v = X[c * NN + n];
    s += v; s2 += v * v;
  }
  float m = s * (1.f / CH);
  float var = s2 * (1.f / CH) - m * m;
  float r = rsqrtf(var + 1e-5f);
  for (int c = 0; c < CH; c += 2) {
    float v0 = (X[c * NN + n] - m) * r * g[c] + be[c];
    float v1 = (X[(c + 1) * NN + n] - m) * r * g[c + 1] + be[c + 1];
    float g0 = 0.5f * v0 * (1.f + erff(v0 * 0.70710678118654752440f));
    float g1 = 0.5f * v1 * (1.f + erff(v1 * 0.70710678118654752440f));
    X2[(c >> 1) * NN + n] = bfpair(g0, g1);
  }
}

// Channel LayerNorm + exact GELU + residual -> f32 (running activation).
__global__ __launch_bounds__(256) void k_ln_gelu_res(const float* __restrict__ X,
                                                     const float* __restrict__ g,
                                                     const float* __restrict__ be,
                                                     const float* __restrict__ res,
                                                     float* __restrict__ Y) {
  int n = blockIdx.x * 256 + threadIdx.x;
  if (n >= NN) return;
  float s = 0.f, s2 = 0.f;
  for (int c = 0; c < CH; ++c) {
    float v = X[c * NN + n];
    s += v; s2 += v * v;
  }
  float m = s * (1.f / CH);
  float var = s2 * (1.f / CH) - m * m;
  float r = rsqrtf(var + 1e-5f);
  for (int c = 0; c < CH; ++c) {
    float v = (X[c * NN + n] - m) * r * g[c] + be[c];
    float ge = 0.5f * v * (1.f + erff(v * 0.70710678118654752440f));
    Y[c * NN + n] = ge + res[c * NN + n];
  }
}

// Y[c][n] = wv[c] * s[n] + bv[c]   (1x1 conv from a 1-channel signal)
__global__ __launch_bounds__(256) void k_outer11(const float* __restrict__ wv,
                                                 const float* __restrict__ bv,
                                                 const float* __restrict__ s,
                                                 float* __restrict__ Y) {
  int i = blockIdx.x * 256 + threadIdx.x;
  if (i >= CH * NN) return;
  int c = i / NN, n = i % NN;
  Y[i] = wv[c] * s[n] + bv[c];
}

__global__ __launch_bounds__(256) void k_add(const float* __restrict__ A,
                                             const float* __restrict__ Bp,
                                             float* __restrict__ Y, int rows) {
  int i = blockIdx.x * 256 + threadIdx.x;
  if (i < rows * NN) Y[i] = A[i] + Bp[i];
}

__global__ __launch_bounds__(256) void k_mask(float* __restrict__ X,
                                              const float* __restrict__ mask,
                                              int rows) {
  int i = blockIdx.x * 256 + threadIdx.x;
  if (i < rows * NN) X[i] *= mask[i % NN];
}

__global__ __launch_bounds__(256) void k_zero(float* __restrict__ p, int n) {
  int i = blockIdx.x * 256 + threadIdx.x;
  if (i < n) p[i] = 0.f;
}

// e = e_q*mask; z_q = (post_ea_m + exp(post_ea_logs)*e)*mask;
// accumulate -0.5*(log2pi+e^2)*mask and mask sums per batch.
__global__ __launch_bounds__(256) void k_prep_post(const float* __restrict__ e_q,
                                                   const float* __restrict__ mask,
                                                   const float* __restrict__ pm,
                                                   const float* __restrict__ plogs,
                                                   float* __restrict__ Z,
                                                   float* accLOGQE, float* accMASK) {
  int n = blockIdx.x * 256 + threadIdx.x;
  int b = n / TT, t = n % TT;
  float msk = mask[n];
  float e0 = e_q[(b * 2 + 0) * TT + t] * msk;
  float e1 = e_q[(b * 2 + 1) * TT + t] * msk;
  Z[0 * NN + n] = (pm[0] + expf(plogs[0]) * e0) * msk;
  Z[1 * NN + n] = (pm[1] + expf(plogs[1]) * e1) * msk;
  float lq = -0.5f * ((C_LOG2PI + e0 * e0) + (C_LOG2PI + e1 * e1)) * msk;
  blockReduceAtomic(lq, accLOGQE + b);
  blockReduceAtomic(msk, accMASK + b);
}

// Rational-quadratic spline forward on channel c1 (per column), h = H[29][N].
// The reference's h*mask is folded into the H loads here (saves a 29xN pass).
__global__ __launch_bounds__(256) void k_rqs(const float* __restrict__ H,
                                             float* __restrict__ Z, int c0, int c1,
                                             const float* __restrict__ mask,
                                             float* __restrict__ ldacc) {
  int n = blockIdx.x * 256 + threadIdx.x;
  int b = n / TT;
  const float scale = 0.072168783648703332f;  // 1/sqrt(192)
  const float min_b = 1e-3f;
  float x_in = Z[c1 * NN + n];
  float msk  = mask[n];

  float uw[KBINS], uh[KBINS], dval[KBINS + 1];
  float uw_max = -1e30f, uh_max = -1e30f;
#pragma unroll
  for (int k = 0; k < KBINS; ++k) {
    uw[k] = H[k * NN + n] * msk * scale;
    uh[k] = H[(KBINS + k) * NN + n] * msk * scale;
    uw_max = fmaxf(uw_max, uw[k]);
    uh_max = fmaxf(uh_max, uh[k]);
  }
  const float padc = logf(expf(1.0f - min_b) - 1.0f);
#pragma unroll
  for (int j = 0; j <= KBINS; ++j) {
    float u = (j == 0 || j == KBINS) ? padc : H[(2 * KBINS + j - 1) * NN + n] * msk;
    float sp = (u > 20.f) ? u : log1pf(expf(u));
    dval[j] = min_b + sp;
  }
  float ws = 0.f, hs = 0.f;
#pragma unroll
  for (int k = 0; k < KBINS; ++k) {
    uw[k] = expf(uw[k] - uw_max); ws += uw[k];
    uh[k] = expf(uh[k] - uh_max); hs += uh[k];
  }
  float cw[KBINS + 1], chh[KBINS + 1];
  cw[0] = -TAILV; chh[0] = -TAILV;
  float cwc = 0.f, chc = 0.f;
  const float coef = 1.f - min_b * KBINS;
#pragma unroll
  for (int k = 0; k < KBINS; ++k) {
    cwc += min_b + coef * uw[k] / ws;
    chc += min_b + coef * uh[k] / hs;
    cw[k + 1]  = 2.f * TAILV * cwc - TAILV;
    chh[k + 1] = 2.f * TAILV * chc - TAILV;
  }
  cw[KBINS] = TAILV; chh[KBINS] = TAILV;

  float x = fminf(fmaxf(x_in, -TAILV), TAILV);
  int idx = -1;
#pragma unroll
  for (int k = 0; k <= KBINS; ++k) idx += (x >= cw[k]) ? 1 : 0;
  idx = min(max(idx, 0), KBINS - 1);

  float in_cw = cw[idx],  in_w = cw[idx + 1] - cw[idx];
  float in_ch = chh[idx], in_h = chh[idx + 1] - chh[idx];
  float in_delta = in_h / in_w;
  float in_d = dval[idx], in_dp1 = dval[idx + 1];
  float theta = (x - in_cw) / in_w;
  float tmt = theta * (1.f - theta);
  float num = in_h * (in_delta * theta * theta + in_d * tmt);
  float den = in_delta + (in_d + in_dp1 - 2.f * in_delta) * tmt;
  float outv = in_ch + num / den;
  float omt = 1.f - theta;
  float dnum = in_delta * in_delta *
               (in_dp1 * theta * theta + 2.f * in_delta * tmt + in_d * omt * omt);
  float lad = logf(dnum) - 2.f * logf(den);
  bool inside = (x_in >= -TAILV) && (x_in <= TAILV);
  float res = inside ? outv : x_in;
  lad = inside ? lad : 0.f;

  Z[c1 * NN + n] = res * msk;
  Z[c0 * NN + n] = Z[c0 * NN + n] * msk;
  blockReduceAtomic(lad * msk, ldacc + b);
}

// sigmoid split + log-sig logdet + log(max(w-u,eps)) + elementwise-affine.
__global__ __launch_bounds__(256) void k_post_final(float* __restrict__ Z,
                                                    int cu, int c1,
                                                    const float* __restrict__ wvec,
                                                    const float* __restrict__ mask,
                                                    const float* __restrict__ em,
                                                    const float* __restrict__ elogs,
                                                    float* accLDQ, float* accLD) {
  int n = blockIdx.x * 256 + threadIdx.x;
  int b = n / TT;
  float msk = mask[n];
  float zu = Z[cu * NN + n];
  float z1 = Z[c1 * NN + n];
  float u  = (1.f / (1.f + expf(-zu))) * msk;
  float z0 = (wvec[n] - u) * msk;
  float lsp = (logsigf(zu) + logsigf(-zu)) * msk;
  float z0l = logf(fmaxf(z0, 1e-5f)) * msk;
  Z[0 * NN + n] = (em[0] + expf(elogs[0]) * z0l) * msk;
  Z[1 * NN + n] = (em[1] + expf(elogs[1]) * z1) * msk;
  blockReduceAtomic(lsp, accLDQ + b);
  blockReduceAtomic(-z0l, accLD + b);
}

__global__ __launch_bounds__(256) void k_nll(const float* __restrict__ Z,
                                             const float* __restrict__ mask,
                                             float* accNLL) {
  int n = blockIdx.x * 256 + threadIdx.x;
  int b = n / TT;
  float msk = mask[n];
  float z0 = Z[0 * NN + n], z1 = Z[1 * NN + n];
  float v = 0.5f * ((C_LOG2PI + z0 * z0) + (C_LOG2PI + z1 * z1)) * msk;
  blockReduceAtomic(v, accNLL + b);
}

__global__ void k_final(float* __restrict__ out, const float* accNLL,
                        const float* accLD, const float* accLOGQE,
                        const float* accLDQ, const float* accMASK,
                        const float* __restrict__ elogs,
                        const float* __restrict__ plogs) {
  int b = threadIdx.x;
  if (b < BB) {
    float ldtot = accLD[b] + (elogs[0] + elogs[1]) * accMASK[b];
    float ldq   = accLDQ[b] + (plogs[0] + plogs[1]) * accMASK[b];
    float logq  = accLOGQE[b] - ldq;
    out[b] = accNLL[b] - ldtot + logq;
  }
}

// ---------------------------------------------------------------------------
extern "C" void kernel_launch(void* const* d_in, const int* in_sizes, int n_in,
                              void* d_out, int out_size, void* d_ws, size_t ws_size,
                              hipStream_t stream) {
  (void)in_sizes; (void)n_in; (void)out_size; (void)ws_size;
  auto P = [&](int i) { return (const float*)d_in[i]; };

  // Param leaf indices (sorted-key pytree flattening of `params`, base = 4).
  const int MAIN_CONVS = 4;       // 3 layers x 8 leaves (b1,b11,b2,g1,g2,sep_b,sep_w,w11)
  const int EA_LOGS = 28, EA_M = 29;
  const int FLOWS = 30;           // 4 x 28 leaves (convs[24], pre_b, pre_w, proj_b, proj_w)
  const int POST_CONVS = 142;
  const int POST_EA_LOGS = 166, POST_EA_M = 167;
  const int POST_FLOWS = 168;
  const int POST_PRE_B = 280, POST_PRE_W = 281;
  const int POST_PROJ_B = 282, POST_PROJ_W = 283;
  const int PRE_B = 284, PRE_W = 285, PROJ_B = 286, PROJ_W = 287;

  const float* x_in = P(0);
  const float* mask = P(1);
  const float* wvec = P(2);
  const float* e_q  = P(3);
  float* out = (float*)d_out;

  // Workspace layout (floats / uints).
  float* ws = (float*)d_ws;
  const size_t SZ = (size_t)CH * NN;
  float* T0  = ws;                 // h_w / g_post
  float* XA  = ws + SZ;            // running activation
  float* TB  = ws + 2 * SZ;
  float* TC  = ws + 3 * SZ;
  float* G   = ws + 4 * SZ;        // x (proj output) = flow conditioning
  float* H29 = ws + 5 * SZ;        // 29(+pad) x N spline params
  float* Z   = H29 + (size_t)32 * NN;
  float* ACC = Z + (size_t)2 * NN;
  float* accLOGQE = ACC, *accMASK = ACC + 32, *accLDQ = ACC + 64;
  float* accLD = ACC + 96, *accNLL = ACC + 128;
  unsigned* BX2 = (unsigned*)(ACC + 192);            // bf16 activations: 96*N u32
  unsigned* APK = BX2 + (size_t)(CH / 2) * NN;       // packed weights: <=12*6*32*8 u32

  const int EB = 256;
  const int gCN = (CH * NN + EB - 1) / EB;           // elementwise over C*N
  const int gHN = ((CH / 2) * NN + EB - 1) / EB;     // pair-packed over (C/2)*N
  const int gN  = NN / EB;                           // per-column kernels

  // GEMM: assumes BX2 already holds the bf16 input activations.
  auto gemm = [&](const float* W, const float* bias, float* Y, int M) {
    int tiles = (M + 15) >> 4;
    k_pack_w<<<(tiles * 6 * 32 + EB - 1) / EB, EB, 0, stream>>>(W, APK, M);
    dim3 grid(NN / (16 * NT), (tiles + 3) / 4);
    if ((M & 15) == 0)
      k_gemm<false><<<grid, 128, 0, stream>>>(APK, BX2, bias, Y, M);
    else
      k_gemm<true><<<grid, 128, 0, stream>>>(APK, BX2, bias, Y, M);
  };

  // dds_conv: in-place on Xa, temps TB/TC; optional additive conditioning g.
  auto dds = [&](float* Xa, int base, const float* gadd) {
    if (gadd) k_add<<<gCN, EB, 0, stream>>>(Xa, gadd, Xa, CH);
    int dil = 1;
    for (int l = 0; l < 3; ++l) {
      const float* b1 = P(base + l * 8 + 0);
      const float* b11 = P(base + l * 8 + 1);
      const float* b2 = P(base + l * 8 + 2);
      const float* g1 = P(base + l * 8 + 3);
      const float* g2 = P(base + l * 8 + 4);
      const float* sep_b = P(base + l * 8 + 5);
      const float* sep_w = P(base + l * 8 + 6);
      const float* w11 = P(base + l * 8 + 7);
      k_dwconv<<<gCN, EB, 0, stream>>>(Xa, mask, sep_w, sep_b, TB, dil);
      k_ln_gelu_pack<<<gN, EB, 0, stream>>>(TB, g1, b1, BX2);
      gemm(w11, b11, TC, CH);
      k_ln_gelu_res<<<gN, EB, 0, stream>>>(TC, g2, b2, Xa, Xa);  // + residual
      dil *= 3;
    }
    k_mask<<<gCN, EB, 0, stream>>>(Xa, mask, CH);
  };

  auto conv_flow = [&](int fb, const float* gbuf, int c0, int c1, float* ldacc) {
    k_outer11<<<gCN, EB, 0, stream>>>(P(fb + 25), P(fb + 24), Z + c0 * NN, XA);
    dds(XA, fb, gbuf);
    k_pack_x<<<gHN, EB, 0, stream>>>(XA, BX2);
    gemm(P(fb + 27), P(fb + 26), H29, 29);
    k_rqs<<<gN, EB, 0, stream>>>(H29, Z, c0, c1, mask, ldacc);
  };

  // --- pipeline ------------------------------------------------------------
  k_zero<<<1, EB, 0, stream>>>(ACC, 160);

  // main conditioning path: x -> pre -> dds -> proj  (= g for flows)
  k_transpose_pack<<<gHN, EB, 0, stream>>>(x_in, BX2);
  gemm(P(PRE_W), P(PRE_B), XA, CH);
  dds(XA, MAIN_CONVS, nullptr);
  k_pack_x<<<gHN, EB, 0, stream>>>(XA, BX2);
  gemm(P(PROJ_W), P(PROJ_B), G, CH);
  k_mask<<<gCN, EB, 0, stream>>>(G, mask, CH);

  // posterior path: w -> post_pre -> dds -> post_proj; g_post = G + h_w
  k_outer11<<<gCN, EB, 0, stream>>>(P(POST_PRE_W), P(POST_PRE_B), wvec, XA);
  dds(XA, POST_CONVS, nullptr);
  k_pack_x<<<gHN, EB, 0, stream>>>(XA, BX2);
  gemm(P(POST_PROJ_W), P(POST_PROJ_B), T0, CH);
  k_mask<<<gCN, EB, 0, stream>>>(T0, mask, CH);
  k_add<<<gCN, EB, 0, stream>>>(G, T0, T0, CH);     // T0 = g_post (G preserved)

  // z_q = post elementwise-affine of e; accumulate logq base + mask sums
  k_prep_post<<<gN, EB, 0, stream>>>(e_q, mask, P(POST_EA_M), P(POST_EA_LOGS),
                                     Z, accLOGQE, accMASK);

  // 4 posterior flows (channel flip between)
  int c0 = 0, c1 = 1;
  for (int f = 0; f < 4; ++f) {
    conv_flow(POST_FLOWS + f * 28, T0, c0, c1, accLDQ);
    int t = c0; c0 = c1; c1 = t;
  }

  // split sigmoid / log transform / elementwise-affine into prior space
  k_post_final<<<gN, EB, 0, stream>>>(Z, c0, c1, wvec, mask,
                                      P(EA_M), P(EA_LOGS), accLDQ, accLD);

  // 4 prior flows conditioned on G
  c0 = 0; c1 = 1;
  for (int f = 0; f < 4; ++f) {
    conv_flow(FLOWS + f * 28, G, c0, c1, accLD);
    int t = c0; c0 = c1; c1 = t;
  }

  // nll + final combine:  out[b] = nll - logdet_tot + logq
  k_nll<<<gN, EB, 0, stream>>>(Z, mask, accNLL);
  k_final<<<1, 32, 0, stream>>>(out, accNLL, accLD, accLOGQE, accLDQ, accMASK,
                                P(EA_LOGS), P(POST_EA_LOGS));
}
